// MaskedMultiHeadAttention_75582834475600
// MI455X (gfx1250) — matrix operand order repaired and checked by reading, not compiled, not run on hardware
//
#include <hip/hip_runtime.h>
#include <hip/hip_bf16.h>

typedef unsigned short u16;
typedef unsigned int   u32;

typedef __attribute__((ext_vector_type(16))) __bf16 v16bf;
typedef __attribute__((ext_vector_type(8)))  __bf16 v8bf;
typedef __attribute__((ext_vector_type(8)))  float  v8f;

#define SLEN 2048
#define DIM  512
#define NH   8
#define TC   128   // t-chunk for online softmax

__device__ __forceinline__ u16 f2bf(float x) {
  u32 u = __float_as_uint(x);
  u += 0x7fffu + ((u >> 16) & 1u);     // round-to-nearest-even
  return (u16)(u >> 16);
}

// ---- WMMA fragment loaders (16-bit, contiguous k = 16*half + s in both A and B) ----
// A (16xM x 32K), source row-major with leading dim ld (elements).
// lane<16: row=lane, holds K = 0..15; lane>=16: row=lane-16, K = 16..31
__device__ __forceinline__ v16bf frag_a(const u16* base, int ld, int row0, int k0) {
  const int lane = threadIdx.x & 31;
  const int half = lane >> 4;
  const int row  = lane & 15;
  const u16* p = base + (size_t)(row0 + row) * ld + k0 + half * 16;
  return *(const v16bf*)(p);
}

// B (32K x 16N) where B(k,n) = R[n0+n, k0+k] of a row-major matrix R (ld elements).
// lane<16: col=lane, K=0..15 ; lane>=16: col=lane-16, K=16..31  (contiguous 32B per lane)
__device__ __forceinline__ v16bf frag_b(const u16* base, int ld, int n0, int k0) {
  const int lane = threadIdx.x & 31;
  const int n  = lane & 15;
  const int kb = (lane >> 4) * 16;
  return *(const v16bf*)(base + (size_t)(n0 + n) * ld + k0 + kb);
}

__device__ __forceinline__ v8f wmma_bf16(v16bf a, v16bf b, v8f c) {
  return __builtin_amdgcn_wmma_f32_16x16x32_bf16(false, a, false, b, (short)0, c, false, false);
}

// ---- Kernel 0: fp32 -> bf16 convert (n multiple of 4) ----
__global__ __launch_bounds__(256) void cvt_kernel(const float4* __restrict__ src,
                                                  uint2* __restrict__ dst, int n4) {
  int i = blockIdx.x * 256 + threadIdx.x;
  if (i < n4) {
    float4 f = src[i];
    uint2 r;
    r.x = (u32)f2bf(f.x) | ((u32)f2bf(f.y) << 16);
    r.y = (u32)f2bf(f.z) | ((u32)f2bf(f.w) << 16);
    dst[i] = r;
  }
}

// ---- Kernel 1: Q/K projection  dst[h,s,e] = X @ W_h^T + b_h  (bf16 out) ----
// one wave = 16(s) x 64(e) tile; tiles = H * 128 * 8 = 8192 -> 1024 blocks x 8 waves
__global__ __launch_bounds__(256) void proj_kernel(const u16* __restrict__ Xb,   // [S, D]
                                                   const u16* __restrict__ Wb,   // [H, D, D] (rows=e)
                                                   const float* __restrict__ bias,// [H, D]
                                                   u16* __restrict__ dst) {      // [H, S, D]
  const int wave = threadIdx.x >> 5;
  const int tile = blockIdx.x * 8 + wave;
  const int et = tile & 7;
  const int st = (tile >> 3) & 127;
  const int h  = tile >> 10;
  const int lane = threadIdx.x & 31;
  const int n = lane & 15, hi = lane >> 4;
  const int s0 = st * 16;
  const u16* W = Wb + (size_t)h * DIM * DIM;

  v8f c[4] = {v8f{}, v8f{}, v8f{}, v8f{}};
  for (int k0 = 0; k0 < DIM; k0 += 32) {
    v16bf a = frag_a(Xb, DIM, s0, k0);
#pragma unroll
    for (int t = 0; t < 4; ++t) {
      v16bf b = frag_b(W, DIM, et * 64 + t * 16, k0);
      c[t] = wmma_bf16(a, b, c[t]);
    }
  }
#pragma unroll
  for (int t = 0; t < 4; ++t) {
    const int e = et * 64 + t * 16 + n;
    const float bval = bias[h * DIM + e];
#pragma unroll
    for (int g = 0; g < 8; ++g) {
      const int s = s0 + g + 8 * hi;
      dst[(size_t)h * SLEN * DIM + (size_t)s * DIM + e] = f2bf(c[t][g] + bval);
    }
  }
}

// ---- Kernel 2: transposed V projection  vT[h,e,t] = (Wv_h @ value^T) + bv ----
// one wave = 16(e) x 64(t) tile; tiles = H * 32 * 32 = 8192 -> 1024 blocks
__global__ __launch_bounds__(256) void projT_kernel(const u16* __restrict__ Wb,   // [H, D, D]
                                                    const u16* __restrict__ Xb,   // value [S, D]
                                                    const float* __restrict__ bias,// [H, D] (per e)
                                                    u16* __restrict__ dst) {      // [H, D, S]
  const int wave = threadIdx.x >> 5;
  const int tile = blockIdx.x * 8 + wave;
  const int tt = tile & 31;
  const int et = (tile >> 5) & 31;
  const int h  = tile >> 10;
  const int lane = threadIdx.x & 31;
  const int n = lane & 15, hi = lane >> 4;
  const u16* W = Wb + (size_t)h * DIM * DIM;

  v8f c[4] = {v8f{}, v8f{}, v8f{}, v8f{}};
  for (int k0 = 0; k0 < DIM; k0 += 32) {
    v16bf a = frag_a(W, DIM, et * 16, k0);     // rows = e
#pragma unroll
    for (int t = 0; t < 4; ++t) {
      v16bf b = frag_b(Xb, DIM, tt * 64 + t * 16, k0);  // cols = t (value rows)
      c[t] = wmma_bf16(a, b, c[t]);
    }
  }
#pragma unroll
  for (int t = 0; t < 4; ++t) {
    const int tcol = tt * 64 + t * 16 + n;
#pragma unroll
    for (int g = 0; g < 8; ++g) {
      const int e = et * 16 + g + 8 * hi;
      dst[(size_t)h * DIM * SLEN + (size_t)e * SLEN + tcol] = f2bf(c[t][g] + bias[h * DIM + e]);
    }
  }
}

// ---- Kernel 3: attention with online softmax ----
// grid (S/16, H), block 256 (8 waves). Each wave owns 64 v-columns of the O accumulator.
__global__ __launch_bounds__(256) void attn_kernel(const u16* __restrict__ qb,   // [H,S,D]
                                                   const u16* __restrict__ kb,   // [H,S,D]
                                                   const u16* __restrict__ vTb,  // [H,D,S]
                                                   const float* __restrict__ mask,// [S,S]
                                                   u16* __restrict__ concat) {   // [S, H*D]
  __shared__ alignas(16) u16   q_lds[16 * DIM];
  __shared__ alignas(16) float sc_lds[16 * TC];
  __shared__ alignas(16) u16   p_lds[16 * TC];
  __shared__ float m_lds[16], f_lds[16], l_lds[16];

  const int h  = blockIdx.y;
  const int s0 = blockIdx.x * 16;
  const int tidx = threadIdx.x;
  const int wave = tidx >> 5, lane = tidx & 31;
  const int n = lane & 15, hi = lane >> 4;
  const float scale = 0.044194173824159216f;  // 1/sqrt(512)

  // stage q block (16 x 512 bf16 = 16 KB) into LDS
  {
    const uint4* src4 = (const uint4*)(qb + (size_t)h * SLEN * DIM + (size_t)s0 * DIM);
    uint4* dst4 = (uint4*)q_lds;
    for (int i = tidx; i < (16 * DIM) / 8; i += 256) dst4[i] = src4[i];
  }
  __syncthreads();

  const u16* kh = kb  + (size_t)h * SLEN * DIM;
  const u16* vh = vTb + (size_t)h * DIM * SLEN;

  float m_run = -3.0e38f, l_run = 0.0f;          // live in wave0 lanes 0..15
  v8f o[4] = {v8f{}, v8f{}, v8f{}, v8f{}};

  for (int t0 = 0; t0 < SLEN; t0 += TC) {
    // 1) one 16x16 score tile per wave (K = 512 over e)
    const int tw = t0 + wave * 16;
    v8f sc = {};
    for (int k0 = 0; k0 < DIM; k0 += 32) {
      v16bf a = frag_a(q_lds, DIM, 0, k0);
      v16bf b = frag_b(kh, DIM, tw, k0);
      sc = wmma_bf16(a, b, sc);
    }
    __syncthreads();   // previous chunk's LDS consumers done
#pragma unroll
    for (int g = 0; g < 8; ++g) {
      const int row = g + 8 * hi;
      const float v = sc[g] * mask[(size_t)(s0 + row) * SLEN + tw + n] * scale;
      sc_lds[row * TC + wave * 16 + n] = v;
    }
    __syncthreads();
    // 2) per-row online-softmax stats (wave 0, one lane per row)
    if (tidx < 16) {
      const int r = tidx;
      float cmax = -3.0e38f;
      for (int j = 0; j < TC; ++j) cmax = fmaxf(cmax, sc_lds[r * TC + j]);
      const float m_new = fmaxf(m_run, cmax);
      const float f = __expf(m_run - m_new);
      float sum = 0.0f;
      for (int j = 0; j < TC; ++j) sum += __expf(sc_lds[r * TC + j] - m_new);
      l_run = l_run * f + sum;
      m_run = m_new;
      m_lds[r] = m_new;
      f_lds[r] = f;
    }
    __syncthreads();
    // 3) P = exp(S - m) as bf16 in LDS
    for (int i = tidx; i < 16 * TC; i += 256) {
      const int r = i / TC;
      p_lds[i] = f2bf(__expf(sc_lds[i] - m_lds[r]));
    }
    __syncthreads();
    // 4) rescale O and accumulate O += P @ vT  (each wave: 4 N-tiles x 4 K-steps)
    float fr[8];
#pragma unroll
    for (int g = 0; g < 8; ++g) fr[g] = f_lds[g + 8 * hi];
#pragma unroll
    for (int t = 0; t < 4; ++t)
#pragma unroll
      for (int g = 0; g < 8; ++g) o[t][g] *= fr[g];
#pragma unroll
    for (int t = 0; t < 4; ++t) {
      const int v0 = wave * 64 + t * 16;
#pragma unroll
      for (int kt = 0; kt < TC / 32; ++kt) {
        v16bf a = frag_a(p_lds, TC, 0, kt * 32);
        v16bf b = frag_b(vh, SLEN, v0, t0 + kt * 32);
        o[t] = wmma_bf16(a, b, o[t]);
      }
    }
  }
  __syncthreads();
  if (tidx < 16) l_lds[tidx] = l_run;
  __syncthreads();

  float inv[8];
#pragma unroll
  for (int g = 0; g < 8; ++g) inv[g] = 1.0f / l_lds[g + 8 * hi];
#pragma unroll
  for (int t = 0; t < 4; ++t) {
    const int col = h * DIM + wave * 64 + t * 16 + n;
#pragma unroll
    for (int g = 0; g < 8; ++g) {
      const int s = s0 + g + 8 * hi;
      concat[(size_t)s * (NH * DIM) + col] = f2bf(o[t][g] * inv[g]);
    }
  }
}

// ---- Kernel 4: output projection  out = concat @ Wo^T + bo  (fp32 out) ----
// one wave = 16(s) x 64(o) tile; tiles = 128 * 8 = 1024 -> 128 blocks x 8 waves
__global__ __launch_bounds__(256) void outproj_kernel(const u16* __restrict__ concat, // [S, H*D]
                                                      const u16* __restrict__ Wob,    // [D, H*D]
                                                      const float* __restrict__ bo,   // [D]
                                                      float* __restrict__ out) {      // [S, D]
  const int wave = threadIdx.x >> 5;
  const int tile = blockIdx.x * 8 + wave;
  const int ot = tile & 7;
  const int st = tile >> 3;
  const int lane = threadIdx.x & 31;
  const int n = lane & 15, hi = lane >> 4;
  const int s0 = st * 16;

  v8f c[4] = {v8f{}, v8f{}, v8f{}, v8f{}};
  for (int k0 = 0; k0 < NH * DIM; k0 += 32) {
    v16bf a = frag_a(concat, NH * DIM, s0, k0);
#pragma unroll
    for (int t = 0; t < 4; ++t) {
      v16bf b = frag_b(Wob, NH * DIM, ot * 64 + t * 16, k0);
      c[t] = wmma_bf16(a, b, c[t]);
    }
  }
#pragma unroll
  for (int t = 0; t < 4; ++t) {
    const int ocol = ot * 64 + t * 16 + n;
    const float bval = bo[ocol];
#pragma unroll
    for (int g = 0; g < 8; ++g) {
      const int s = s0 + g + 8 * hi;
      out[(size_t)s * DIM + ocol] = c[t][g] + bval;
    }
  }
}

extern "C" void kernel_launch(void* const* d_in, const int* in_sizes, int n_in,
                              void* d_out, int out_size, void* d_ws, size_t ws_size,
                              hipStream_t stream) {
  (void)in_sizes; (void)n_in; (void)out_size; (void)ws_size;
  const float* query = (const float*)d_in[0];
  const float* key_  = (const float*)d_in[1];
  const float* value = (const float*)d_in[2];
  const float* mask  = (const float*)d_in[3];
  const float* Wq = (const float*)d_in[4];
  const float* bq = (const float*)d_in[5];
  const float* Wk = (const float*)d_in[6];
  const float* bk = (const float*)d_in[7];
  const float* Wv = (const float*)d_in[8];
  const float* bv = (const float*)d_in[9];
  const float* Wo = (const float*)d_in[10];
  const float* bo = (const float*)d_in[11];
  float* out = (float*)d_out;

  char* ws = (char*)d_ws;
  const size_t MB = 1024ull * 1024ull;
  u16* qb     = (u16*)(ws + 0 * MB);     // 16 MB  [H,S,D]
  u16* kb     = (u16*)(ws + 16 * MB);    // 16 MB  [H,S,D]
  u16* vTb    = (u16*)(ws + 32 * MB);    // 16 MB  [H,D,S]
  u16* concat = (u16*)(ws + 48 * MB);    // 16 MB  [S,H*D]
  u16* Xq  = (u16*)(ws + 64 * MB);       // 2 MB
  u16* Xk  = (u16*)(ws + 66 * MB);       // 2 MB
  u16* Xv  = (u16*)(ws + 68 * MB);       // 2 MB
  u16* Wqb = (u16*)(ws + 70 * MB);       // 4 MB
  u16* Wkb = (u16*)(ws + 74 * MB);       // 4 MB
  u16* Wvb = (u16*)(ws + 78 * MB);       // 4 MB
  u16* Wob = (u16*)(ws + 82 * MB);       // 4 MB

  const int nX4 = SLEN * DIM / 4;            // 262144
  const int nW4 = NH * DIM * DIM / 4;        // 524288
  const int nWo4 = DIM * NH * DIM / 4;       // 524288

  cvt_kernel<<<(nX4 + 255) / 256, 256, 0, stream>>>((const float4*)query, (uint2*)Xq, nX4);
  cvt_kernel<<<(nX4 + 255) / 256, 256, 0, stream>>>((const float4*)key_,  (uint2*)Xk, nX4);
  cvt_kernel<<<(nX4 + 255) / 256, 256, 0, stream>>>((const float4*)value, (uint2*)Xv, nX4);
  cvt_kernel<<<(nW4 + 255) / 256, 256, 0, stream>>>((const float4*)Wq, (uint2*)Wqb, nW4);
  cvt_kernel<<<(nW4 + 255) / 256, 256, 0, stream>>>((const float4*)Wk, (uint2*)Wkb, nW4);
  cvt_kernel<<<(nW4 + 255) / 256, 256, 0, stream>>>((const float4*)Wv, (uint2*)Wvb, nW4);
  cvt_kernel<<<(nWo4 + 255) / 256, 256, 0, stream>>>((const float4*)Wo, (uint2*)Wob, nWo4);

  proj_kernel<<<1024, 256, 0, stream>>>(Xq, Wqb, bq, qb);
  proj_kernel<<<1024, 256, 0, stream>>>(Xk, Wkb, bk, kb);
  projT_kernel<<<1024, 256, 0, stream>>>(Wvb, Xv, bv, vTb);

  attn_kernel<<<dim3(SLEN / 16, NH), 256, 0, stream>>>(qb, kb, vTb, mask, concat);

  outproj_kernel<<<128, 256, 0, stream>>>(concat, Wob, bo, out);
}
